// SparseResBlock_16389595201741
// MI455X (gfx1250) — compile-verified, matrix operand order checked by reading, not run on hardware
//
#include <hip/hip_runtime.h>
#include <hip/hip_bf16.h>

// ---------------- problem constants (match reference) ----------------
#define GDIM   128
#define BDIM   2
#define TABLE_SZ (BDIM * GDIM * GDIM * GDIM)   // 4,194,304
#define NOFF   27
#define CIN    64
#define COUT   64
#define EPSBN  1e-5f

typedef __attribute__((ext_vector_type(16))) __bf16 v16bf;
typedef __attribute__((ext_vector_type(8)))  float  v8f;

// channel permutation so a lane's 16 A-fragment values are contiguous:
// within each 32-channel K-slice: [0..7,16..23 | 8..15,24..31]
__device__ __forceinline__ int chan_perm(int c) {
    int s = c >> 5;          // K-slice
    int r = c & 31;
    int h = (r >> 3) & 1;    // 8..15 / 24..31 -> second half (lanes 16-31)
    int g = r >> 4;          // 0..15 vs 16..31
    return s * 32 + h * 16 + g * 8 + (r & 7);
}

// ---------------- neighbor construction ----------------
__global__ void k_init_table(int* __restrict__ table) {
    int i = blockIdx.x * blockDim.x + threadIdx.x;
    if (i < TABLE_SZ) table[i] = -1;
}

__global__ void k_scatter(const int* __restrict__ coords, int* __restrict__ table, int n) {
    int i = blockIdx.x * blockDim.x + threadIdx.x;
    if (i >= n) return;
    int b = coords[i * 4 + 0], x = coords[i * 4 + 1];
    int y = coords[i * 4 + 2], z = coords[i * 4 + 3];
    int lin = ((b * GDIM + x) * GDIM + y) * GDIM + z;
    table[lin] = i;
}

__global__ void k_neighbors(const int* __restrict__ coords, const int* __restrict__ table,
                            int* __restrict__ nbr, int n) {
    int i = blockIdx.x * blockDim.x + threadIdx.x;
    if (i >= n) return;
    int b = coords[i * 4 + 0], x = coords[i * 4 + 1];
    int y = coords[i * 4 + 2], z = coords[i * 4 + 3];
    #pragma unroll
    for (int k = 0; k < NOFF; ++k) {
        int dx = k / 9 - 1, dy = (k / 3) % 3 - 1, dz = k % 3 - 1;
        int nx = x + dx, ny = y + dy, nz = z + dz;
        int idx = -1;
        if ((unsigned)nx < GDIM && (unsigned)ny < GDIM && (unsigned)nz < GDIM) {
            int lin = ((b * GDIM + nx) * GDIM + ny) * GDIM + nz;
            idx = table[lin];
        }
        nbr[(size_t)k * n + i] = idx;
    }
}

// ---------------- packing kernels ----------------
__global__ void k_pack_feats(const float* __restrict__ f, __hip_bfloat16* __restrict__ fpk,
                             long total) {
    long i = (long)blockIdx.x * blockDim.x + threadIdx.x;
    if (i >= total) return;
    int c = (int)(i & 63);
    long r = i >> 6;
    fpk[r * 64 + chan_perm(c)] = __float2bfloat16(f[i]);
}

// wpk layout: [k(27)][s(2)][t(4)][lane(32)][j(16)] bf16  (1 KB per fragment)
__global__ void k_pack_weights(const float* __restrict__ w, __hip_bfloat16* __restrict__ wpk) {
    int i = blockIdx.x * blockDim.x + threadIdx.x;
    const int TOT = NOFF * 2 * 4 * 32 * 16;
    if (i >= TOT) return;
    int j    = i & 15;
    int lane = (i >> 4) & 31;
    int t    = (i >> 9) & 3;
    int s    = (i >> 11) & 1;
    int k    = i >> 12;
    int ncol = lane & 15;
    int h    = lane >> 4;
    int kk   = (j < 8 ? j : j + 8) + h * 8;     // K within 32 per B-fragment layout
    int cin  = s * 32 + kk;
    int cout = t * 16 + ncol;
    wpk[i] = __float2bfloat16(w[((size_t)k * CIN + cin) * COUT + cout]);
}

// ---------------- gather-GEMM conv (WMMA, 32-voxel M-tile per wave) ----------------
#define W_FRAG_ELEMS 512                         // 32 lanes * 16 bf16
#define W_TOTAL_BYTES (NOFF * 2 * 4 * W_FRAG_ELEMS * 2)  // 221184 B

__global__ void __launch_bounds__(256)
k_conv_wmma(const __hip_bfloat16* __restrict__ fpk,
            const __hip_bfloat16* __restrict__ wpk,
            const int* __restrict__ nbr,
            float* __restrict__ out, int n) {
    extern __shared__ __hip_bfloat16 lw[];       // 216 KB packed weights

    // async DMA stage of all weight fragments into LDS (ASYNCcnt-tracked path)
    {
        const int nvec = W_TOTAL_BYTES / 16;
        unsigned ldsbase = (unsigned)(size_t)lw;
        for (int i = threadIdx.x; i < nvec; i += blockDim.x) {
            unsigned lds_off = ldsbase + (unsigned)i * 16u;
            unsigned long long ga =
                (unsigned long long)wpk + (unsigned long long)i * 16ull;
            asm volatile("global_load_async_to_lds_b128 %0, %1, off"
                         :: "v"(lds_off), "v"(ga) : "memory");
        }
        asm volatile("s_wait_asynccnt 0" ::: "memory");
    }
    __syncthreads();

    const int lane = threadIdx.x & 31;
    const int wave = threadIdx.x >> 5;
    const int wpb  = blockDim.x >> 5;
    const int m    = lane & 15;
    const int half = lane >> 4;                  // A K-half selector
    const int nn   = lane & 15;                  // D column within 16
    const int tp_count = (n + 31) >> 5;          // 32-voxel tile pairs

    for (int tp = blockIdx.x * wpb + wave; tp < tp_count; tp += gridDim.x * wpb) {
        const int r0 = tp * 32 + m;
        const int r1 = r0 + 16;
        const bool v0 = r0 < n, v1 = r1 < n;

        v8f acc[8] = {};

        for (int k = 0; k < NOFF; ++k) {
            int i0 = v0 ? nbr[(size_t)k * n + r0] : -1;
            int i1 = v1 ? nbr[(size_t)k * n + r1] : -1;
            v16bf a00 = {}, a01 = {}, a10 = {}, a11 = {};
            if (i0 >= 0) {
                const v16bf* p = (const v16bf*)(fpk + (size_t)i0 * 64 + half * 16);
                a00 = p[0];          // K-slice 0 (cin 0..31)
                a01 = p[2];          // K-slice 1 (cin 32..63)
            }
            if (i1 >= 0) {
                const v16bf* p = (const v16bf*)(fpk + (size_t)i1 * 64 + half * 16);
                a10 = p[0];
                a11 = p[2];
            }
            const __hip_bfloat16* fb =
                lw + (size_t)(k * 8) * W_FRAG_ELEMS + (size_t)lane * 16;
            #pragma unroll
            for (int t = 0; t < 4; ++t) {
                v16bf b0 = *(const v16bf*)(fb + (size_t)t * W_FRAG_ELEMS);
                v16bf b1 = *(const v16bf*)(fb + (size_t)(4 + t) * W_FRAG_ELEMS);
                acc[t]     = __builtin_amdgcn_wmma_f32_16x16x32_bf16(
                    false, a00, false, b0, (short)0, acc[t],     false, false);
                acc[t]     = __builtin_amdgcn_wmma_f32_16x16x32_bf16(
                    false, a01, false, b1, (short)0, acc[t],     false, false);
                acc[4 + t] = __builtin_amdgcn_wmma_f32_16x16x32_bf16(
                    false, a10, false, b0, (short)0, acc[4 + t], false, false);
                acc[4 + t] = __builtin_amdgcn_wmma_f32_16x16x32_bf16(
                    false, a11, false, b1, (short)0, acc[4 + t], false, false);
            }
        }

        // C/D layout: VGPR j -> M = j + half*8, N = (lane&15) + t*16
        #pragma unroll
        for (int t = 0; t < 4; ++t) {
            #pragma unroll
            for (int j = 0; j < 8; ++j) {
                int rm0 = tp * 32 + half * 8 + j;
                int rm1 = rm0 + 16;
                if (rm0 < n) out[(size_t)rm0 * 64 + t * 16 + nn] = acc[t][j];
                if (rm1 < n) out[(size_t)rm1 * 64 + t * 16 + nn] = acc[4 + t][j];
            }
        }
    }
}

// ---------------- BN stats (deterministic two-stage) ----------------
#define RED_BLOCKS 512

__global__ void k_reduce_stats(const float* __restrict__ x, float* __restrict__ partials, int n) {
    __shared__ float ssum[256], ssq[256];
    int c   = threadIdx.x & 63;
    int sub = threadIdx.x >> 6;                  // 0..3 row streams per block
    float s = 0.f, q = 0.f;
    for (int r = blockIdx.x * 4 + sub; r < n; r += gridDim.x * 4) {
        float v = x[(size_t)r * 64 + c];
        s += v; q += v * v;
    }
    ssum[threadIdx.x] = s; ssq[threadIdx.x] = q;
    __syncthreads();
    if (threadIdx.x < 64) {
        float S = 0.f, Q = 0.f;
        #pragma unroll
        for (int i = 0; i < 4; ++i) { S += ssum[threadIdx.x + 64 * i]; Q += ssq[threadIdx.x + 64 * i]; }
        partials[(size_t)blockIdx.x * 128 + threadIdx.x]      = S;
        partials[(size_t)blockIdx.x * 128 + 64 + threadIdx.x] = Q;
    }
}

__global__ void k_finalize_stats(const float* __restrict__ partials,
                                 const float* __restrict__ gamma, const float* __restrict__ beta,
                                 float* __restrict__ ss, int n) {
    int c = threadIdx.x;
    if (c >= 64) return;
    float S = 0.f, Q = 0.f;
    for (int b = 0; b < RED_BLOCKS; ++b) {
        S += partials[(size_t)b * 128 + c];
        Q += partials[(size_t)b * 128 + 64 + c];
    }
    float mean = S / n;
    float var  = Q / n - mean * mean;
    float sc   = gamma[c] * rsqrtf(var + EPSBN);
    ss[c]      = sc;
    ss[64 + c] = beta[c] - mean * sc;
}

// ---------------- elementwise epilogues ----------------
__global__ void k_bnrelu_pack(const float* __restrict__ x, const float* __restrict__ ss,
                              __hip_bfloat16* __restrict__ hpk, long total) {
    long i = (long)blockIdx.x * blockDim.x + threadIdx.x;
    if (i >= total) return;
    int c = (int)(i & 63);
    long r = i >> 6;
    float v = fmaxf(x[i] * ss[c] + ss[64 + c], 0.f);
    hpk[r * 64 + chan_perm(c)] = __float2bfloat16(v);
}

__global__ void k_bn_add_relu(float* __restrict__ x, const float* __restrict__ ss,
                              const float* __restrict__ feats, long total) {
    long i = (long)blockIdx.x * blockDim.x + threadIdx.x;
    if (i >= total) return;
    int c = (int)(i & 63);
    float v = x[i] * ss[c] + ss[64 + c] + feats[i];
    x[i] = fmaxf(v, 0.f);
}

// ---------------- host launch ----------------
extern "C" void kernel_launch(void* const* d_in, const int* in_sizes, int n_in,
                              void* d_out, int out_size, void* d_ws, size_t ws_size,
                              hipStream_t stream) {
    const float* feats  = (const float*)d_in[0];
    const float* w1     = (const float*)d_in[1];
    const float* g1     = (const float*)d_in[2];
    const float* b1     = (const float*)d_in[3];
    const float* w2     = (const float*)d_in[4];
    const float* g2     = (const float*)d_in[5];
    const float* b2     = (const float*)d_in[6];
    const int*   coords = (const int*)d_in[7];
    float* out = (float*)d_out;

    const int  n     = in_sizes[0] / 64;
    const long total = (long)n * 64;

    // workspace layout (256 B aligned)
    size_t off = 0;
    auto carve = [&](size_t bytes) { size_t o = off; off += (bytes + 255) & ~(size_t)255; return o; };
    char* ws = (char*)d_ws;
    int*             table    = (int*)(ws + carve((size_t)TABLE_SZ * 4));
    int*             nbr      = (int*)(ws + carve((size_t)NOFF * n * 4));
    __hip_bfloat16*  fpk      = (__hip_bfloat16*)(ws + carve((size_t)total * 2));
    __hip_bfloat16*  hpk      = (__hip_bfloat16*)(ws + carve((size_t)total * 2));
    __hip_bfloat16*  wpk1     = (__hip_bfloat16*)(ws + carve((size_t)W_TOTAL_BYTES));
    __hip_bfloat16*  wpk2     = (__hip_bfloat16*)(ws + carve((size_t)W_TOTAL_BYTES));
    float*           partials = (float*)(ws + carve((size_t)RED_BLOCKS * 128 * 4));
    float*           ss1      = (float*)(ws + carve(128 * 4));
    float*           ss2      = (float*)(ws + carve(128 * 4));
    (void)ws_size; (void)n_in; (void)out_size;

    const int T = 256;
    const int WPACK_TOT = NOFF * 2 * 4 * 32 * 16;

    // neighbor map
    k_init_table<<<(TABLE_SZ + T - 1) / T, T, 0, stream>>>(table);
    k_scatter<<<(n + T - 1) / T, T, 0, stream>>>(coords, table, n);
    k_neighbors<<<(n + T - 1) / T, T, 0, stream>>>(coords, table, nbr, n);

    // packing
    k_pack_weights<<<(WPACK_TOT + T - 1) / T, T, 0, stream>>>(w1, wpk1);
    k_pack_weights<<<(WPACK_TOT + T - 1) / T, T, 0, stream>>>(w2, wpk2);
    k_pack_feats<<<(int)((total + T - 1) / T), T, 0, stream>>>(feats, fpk, total);

    const int convBlocks = 512;
    const int ewBlocks   = (int)((total + T - 1) / T);

    // conv1 -> d_out ; BN1 + ReLU -> hpk (bf16, packed)
    k_conv_wmma<<<convBlocks, T, W_TOTAL_BYTES, stream>>>(fpk, wpk1, nbr, out, n);
    k_reduce_stats<<<RED_BLOCKS, T, 0, stream>>>(out, partials, n);
    k_finalize_stats<<<1, 64, 0, stream>>>(partials, g1, b1, ss1, n);
    k_bnrelu_pack<<<ewBlocks, T, 0, stream>>>(out, ss1, hpk, total);

    // conv2 -> d_out ; BN2 + residual + ReLU in place
    k_conv_wmma<<<convBlocks, T, W_TOTAL_BYTES, stream>>>(hpk, wpk2, nbr, out, n);
    k_reduce_stats<<<RED_BLOCKS, T, 0, stream>>>(out, partials, n);
    k_finalize_stats<<<1, 64, 0, stream>>>(partials, g2, b2, ss2, n);
    k_bn_add_relu<<<ewBlocks, T, 0, stream>>>(out, ss2, feats, total);
}
